// DSDBlock_7370163880330
// MI455X (gfx1250) — matrix operand order repaired and checked by reading, not compiled, not run on hardware
//
#include <hip/hip_runtime.h>
#include <cmath>

// ---------------------------------------------------------------------------
// DSD block, MI455X (gfx1250, wave32, WGP, 320KB LDS, WMMA).
// B=32, T=512, C=128, K=3, periods p in [8,64], L=CYC_MAX=64, P_MAX=64.
//
// Simplification used (exact): unfold(fold(x)) == x for t<T, so
//   y[b,k,t,c]  = x[b,t,c] + gamma[c]*hh[n,c,t/p_k]
//   out[b,t,c]  = x - Gamma_b*(num_c/den_c)*(x+s),  s = sum_k w_k*gamma*hh
// with num_c = sum_t (-s)(x+s), den_c = sum_t (x+s)^2 + 1e-6.
// ---------------------------------------------------------------------------

#define B_   32
#define T_   512
#define C_   128
#define K_   3
#define LMAX 64
#define KW_  9

typedef __attribute__((ext_vector_type(16))) __bf16 v16bf;
typedef __attribute__((ext_vector_type(8)))  float  v8f;

__device__ __forceinline__ float wave_sum32(float v) {
#pragma unroll
  for (int o = 16; o > 0; o >>= 1) v += __shfl_down(v, o, 32);
  return v;
}

// ---------------------------------------------------------------------------
// Kernel 1: circular autocorrelation (lags 8..64), top-3, softmax, entropy.
// One block per batch b; x[b] (256KB) staged entirely in LDS (fits in 320KB).
// 8 waves x 8 passes cover 64 lag slots; per-lag dot over 512x128 from LDS.
// ---------------------------------------------------------------------------
__global__ void k_autocorr(const float* __restrict__ x,
                           int* __restrict__ p_arr,
                           float* __restrict__ w_arr,
                           float* __restrict__ g_arr) {
  extern __shared__ char smem_raw[];
  float* xs = (float*)smem_raw;     // 512*128 floats = 256KB
  float* rs = xs + T_ * C_;         // 64 floats
  const int b   = blockIdx.x;
  const int tid = threadIdx.x;      // 256 threads, 8 waves
  const float* xb = x + (size_t)b * T_ * C_;
  for (int i = tid; i < T_ * C_; i += 256) xs[i] = xb[i];
  __syncthreads();

  const int lane = tid & 31;
  const int wv   = tid >> 5;        // 0..7
  for (int pass = 0; pass < 8; ++pass) {
    int slot = pass * 8 + wv;       // 0..63
    int lag  = 8 + slot;
    float acc = 0.f;
    if (lag <= 64) {
      for (int t = 0; t < T_; ++t) {
        int t2 = t + lag; if (t2 >= T_) t2 -= T_;
        const float* r0 = xs + t * C_;
        const float* r1 = xs + t2 * C_;
#pragma unroll
        for (int cc = 0; cc < 4; ++cc) {
          int c = cc * 32 + lane;
          acc = fmaf(r0[c], r1[c], acc);
        }
      }
    }
    acc = wave_sum32(acc);
    if (lane == 0 && lag <= 64) rs[slot] = acc * (1.0f / 128.0f);  // mean over C
  }
  __syncthreads();

  if (tid == 0) {
    float bv[3]; int bi[3];
    for (int k = 0; k < 3; ++k) {
      float mv = -3.4e38f; int mi = 0;
      for (int l = 0; l < 57; ++l) {          // lags 8..64
        bool used = false;
        for (int m = 0; m < k; ++m) if (bi[m] == l) used = true;
        if (!used && rs[l] > mv) { mv = rs[l]; mi = l; }
      }
      bv[k] = mv; bi[k] = mi;
    }
    float mx = bv[0];
    float e0 = expf(bv[0] - mx), e1 = expf(bv[1] - mx), e2 = expf(bv[2] - mx);
    float den = e0 + e1 + e2;
    float w0 = e0 / den, w1 = e1 / den, w2 = e2 / den;
    float H = -(w0 * logf(w0 + 1e-8f) + w1 * logf(w1 + 1e-8f) + w2 * logf(w2 + 1e-8f));
    float dl = logf(3.0f + 1e-8f) + 1e-8f;
    float G = 1.0f - H / dl;
    G = fminf(fmaxf(G, 0.0f), 1.0f);
    p_arr[b * 3 + 0] = 8 + bi[0];
    p_arr[b * 3 + 1] = 8 + bi[1];
    p_arr[b * 3 + 2] = 8 + bi[2];
    w_arr[b * 3 + 0] = w0; w_arr[b * 3 + 1] = w1; w_arr[b * 3 + 2] = w2;
    g_arr[b] = G;
  }
}

// ---------------------------------------------------------------------------
// Kernel 2: per-n (=b*K+k) fold-mean -> depthwise conv -> WMMA pointwise
// GEMM H = W_pw(128x128) @ hd(128x64) in bf16 -> GroupNorm -> GELU -> gate
// -> A[n,c,l] = gamma[c]*hh[c,l].  One block (1024 thr = 32 waves) per n.
// 32 waves own the 8x4 grid of 16x16 output tiles; K=128 in 4 chunks of 32.
// Fragments per ISA 7.12.2 (bank-conflict-padded LDS strides 65/66/130).
// ---------------------------------------------------------------------------
__global__ void k_cycconv(const float* __restrict__ x,
                          const float* __restrict__ Wdw,
                          const float* __restrict__ Wpw,
                          const float* __restrict__ gnw,
                          const float* __restrict__ gnb,
                          const float* __restrict__ Wg,
                          const float* __restrict__ gamma,
                          const int* __restrict__ p_arr,
                          float* __restrict__ Aout) {
  extern __shared__ char smem_raw[];
  float*  u_s    = (float*)smem_raw;        // 128*65
  float*  H_s    = u_s + 128 * 65;          // 128*65
  float*  ubar_s = H_s + 128 * 65;          // 128
  float*  gate_s = ubar_s + 128;            // 128
  float*  mean_s = gate_s + 128;            // 32
  float*  rstd_s = mean_s + 32;             // 32
  __bf16* hd_s   = (__bf16*)(rstd_s + 32);  // 128*66
  __bf16* pwb_s  = hd_s + 128 * 66;         // 128*130

  const int n    = blockIdx.x;              // 0..95
  const int tid  = threadIdx.x;             // 1024
  const int b    = n / 3;
  const int p    = p_arr[n];
  const int ncyc = (T_ + p - 1) / p;
  const float* xb = x + (size_t)b * T_ * C_;

  // stage W_pw as bf16 into LDS (row=o, col=c, stride 130 to dodge conflicts)
  for (int idx = tid; idx < C_ * C_; idx += 1024) {
    int o = idx >> 7, ci = idx & 127;
    pwb_s[o * 130 + ci] = (__bf16)Wpw[idx];
  }

  // Phase A: u[c,l] = (1/64) * sum_{pos<p} x_reflect[b, l*p+pos, c]
  {
    const int c  = tid & 127;   // lanes -> consecutive c: coalesced global reads
    const int lg = tid >> 7;    // 0..7
#pragma unroll
    for (int i = 0; i < 8; ++i) {
      int l = lg * 8 + i;
      float acc = 0.f;
      if (l < ncyc) {
        int base = l * p;
        for (int pos = 0; pos < p; ++pos) {
          int time = base + pos;
          int src = (time < T_) ? time : (2 * (T_ - 1) - time);  // reflect
          acc += xb[src * C_ + c];
        }
      }
      u_s[c * 65 + l] = acc * (1.0f / 64.0f);
    }
  }
  __syncthreads();

  // Phase B: ubar[c] = mean_l u[c,l]
  if (tid < C_) {
    float s = 0.f;
#pragma unroll
    for (int l = 0; l < 64; ++l) s += u_s[tid * 65 + l];
    ubar_s[tid] = s * (1.0f / 64.0f);
  }
  // Phase C: depthwise conv (KW=9, pad 4), emit bf16 hd[c,l] (stride 66)
  {
    const int c  = tid & 127;
    const int lg = tid >> 7;
    float wv[KW_];
#pragma unroll
    for (int j = 0; j < KW_; ++j) wv[j] = Wdw[c * KW_ + j];
#pragma unroll
    for (int i = 0; i < 8; ++i) {
      int l = lg * 8 + i;
      float acc = 0.f;
#pragma unroll
      for (int j = 0; j < KW_; ++j) {
        int ll = l + j - 4;
        if (ll >= 0 && ll < 64) acc += wv[j] * u_s[c * 65 + ll];
      }
      hd_s[c * 66 + l] = (__bf16)acc;
    }
  }
  __syncthreads();

  // Phase D: gate[o] = sigmoid(W_gate[o,:] . ubar)
  if (tid < C_) {
    float z = 0.f;
    for (int c = 0; c < C_; ++c) z += Wg[tid * C_ + c] * ubar_s[c];
    gate_s[tid] = 1.0f / (1.0f + expf(-z));
  }

  // Phase E: WMMA  H(128x64) = W_pw(128x128) @ hd(128x64), bf16 in, f32 acc.
  // All 1024 threads converged here (EXEC must be all 1s for WMMA).
  {
    const int lane  = tid & 31;
    const int wv    = tid >> 5;    // 0..31 : one 16x16 tile per wave
    const int mbase = (wv >> 2) * 16;   // 8 M-tiles (o)
    const int nbase = (wv & 3) * 16;    // 4 N-tiles (l)
    const int half  = lane >> 4;
    const int mrow  = mbase + (lane & 15);
    v8f acc = {};
#pragma unroll
    for (int kc = 0; kc < 4; ++kc) {
      int kbase = kc * 32;
      v16bf afrag, bfrag;
      // A frag (16x32 bf16): lane = M; K = v + 8*half + (v>=8 ? 8 : 0)
#pragma unroll
      for (int v = 0; v < 16; ++v) {
        int kk = kbase + v + 8 * half + ((v >= 8) ? 8 : 0);
        afrag[v] = pwb_s[mrow * 130 + kk];
      }
      // B frag (32x16 bf16): lane = N; K = v + 8*half + (v>=8 ? 8 : 0)
      int bcol = nbase + (lane & 15);
#pragma unroll
      for (int v = 0; v < 16; ++v)
        bfrag[v] = hd_s[(kbase + v + 8 * half + ((v >= 8) ? 8 : 0)) * 66 + bcol];
      acc = __builtin_amdgcn_wmma_f32_16x16x32_bf16(false, afrag, false, bfrag,
                                                    (short)0, acc, false, false);
    }
    // D frag (16x16 f32): lane = N (lane&15); VGPR r -> M = r + 8*half
#pragma unroll
    for (int r = 0; r < 8; ++r) {
      int m  = mbase + r + 8 * half;
      int nn = nbase + (lane & 15);
      H_s[m * 65 + nn] = acc[r];
    }
  }
  __syncthreads();

  // Phase F1: GroupNorm stats, 32 groups of (4c x 64l) = 256 vals, 1 wave each
  {
    const int gi   = tid >> 5;
    const int lane = tid & 31;
    float s1 = 0.f, s2 = 0.f;
#pragma unroll
    for (int i = 0; i < 8; ++i) {
      int idx = lane + 32 * i;          // 0..255
      float v = H_s[(gi * 4 + (idx >> 6)) * 65 + (idx & 63)];
      s1 += v; s2 += v * v;
    }
    s1 = wave_sum32(s1);
    s2 = wave_sum32(s2);
    if (lane == 0) {
      float mean = s1 * (1.0f / 256.0f);
      float var  = s2 * (1.0f / 256.0f) - mean * mean;
      mean_s[gi] = mean;
      rstd_s[gi] = rsqrtf(var + 1e-5f);
    }
  }
  __syncthreads();

  // Phase F2: GN affine + exact GELU + gate + gamma -> Aout[n][c][l]
  {
    float* An = Aout + (size_t)n * C_ * LMAX;
    const int l  = tid & 63;
    const int c0 = tid >> 6;            // 0..15
#pragma unroll
    for (int it = 0; it < 8; ++it) {
      int c  = c0 + 16 * it;
      int gi = c >> 2;
      float h  = H_s[c * 65 + l];
      float hn = (h - mean_s[gi]) * rstd_s[gi] * gnw[c] + gnb[c];
      float ge = 0.5f * hn * (1.0f + erff(hn * 0.70710678118654752f));
      An[c * 64 + l] = ge * gate_s[c] * gamma[c];
    }
  }
}

// ---------------------------------------------------------------------------
// Kernel 3: combine. One block per b. A tiles (pre-scaled by w_k) in LDS,
// two deterministic passes over t: (num,den) per channel, then the blend.
// ---------------------------------------------------------------------------
__global__ void k_combine(const float* __restrict__ x,
                          const float* __restrict__ Aarr,
                          const int* __restrict__ p_arr,
                          const float* __restrict__ w_arr,
                          const float* __restrict__ g_arr,
                          float* __restrict__ out) {
  extern __shared__ char smem_raw[];
  float* As     = (float*)smem_raw;            // 3*128*65 (stride 65: no conflicts)
  float* partN  = As + 3 * 128 * 65;           // 8*128
  float* partD  = partN + 8 * 128;             // 8*128
  float* coef_s = partD + 8 * 128;             // 128

  const int b   = blockIdx.x;
  const int tid = threadIdx.x;                 // 1024
  const int p0 = p_arr[b * 3 + 0], p1 = p_arr[b * 3 + 1], p2 = p_arr[b * 3 + 2];
  const float w0 = w_arr[b * 3 + 0], w1 = w_arr[b * 3 + 1], w2 = w_arr[b * 3 + 2];
  const float g  = g_arr[b];
  const float* xb = x + (size_t)b * T_ * C_;
  float* ob = out + (size_t)b * T_ * C_;

  // load A tiles for the 3 candidate periods, pre-scaled by softmax weight
  for (int idx = tid; idx < 3 * C_ * LMAX; idx += 1024) {
    int k   = idx / (C_ * LMAX);
    int rem = idx - k * (C_ * LMAX);
    int c = rem >> 6, l = rem & 63;
    float wk = (k == 0) ? w0 : ((k == 1) ? w1 : w2);
    As[k * (128 * 65) + c * 65 + l] =
        wk * Aarr[(size_t)(b * 3 + k) * C_ * LMAX + rem];
  }
  __syncthreads();

  const int c  = tid & 127;     // lanes -> consecutive c: coalesced x reads
  const int tg = tid >> 7;      // 0..7, 64 t's each
  float pn = 0.f, pd = 0.f;
  for (int i = 0; i < 64; ++i) {
    int t = tg * 64 + i;
    float s = As[0 * (128 * 65) + c * 65 + (t / p0)]
            + As[1 * (128 * 65) + c * 65 + (t / p1)]
            + As[2 * (128 * 65) + c * 65 + (t / p2)];
    float per = xb[t * C_ + c] + s;     // periodic
    pn -= s * per;                      // local = -s
    pd += per * per;
  }
  partN[tg * 128 + c] = pn;             // deterministic staged reduction
  partD[tg * 128 + c] = pd;
  __syncthreads();
  if (tid < C_) {
    float sn = 0.f, sd = 0.f;
#pragma unroll
    for (int j = 0; j < 8; ++j) { sn += partN[j * 128 + tid]; sd += partD[j * 128 + tid]; }
    coef_s[tid] = g * sn / (sd + 1e-6f);
  }
  __syncthreads();
  for (int i = 0; i < 64; ++i) {
    int t = tg * 64 + i;
    float s = As[0 * (128 * 65) + c * 65 + (t / p0)]
            + As[1 * (128 * 65) + c * 65 + (t / p1)]
            + As[2 * (128 * 65) + c * 65 + (t / p2)];
    float xv = xb[t * C_ + c];
    ob[t * C_ + c] = xv - coef_s[c] * (xv + s);
  }
}

// ---------------------------------------------------------------------------
// Host launcher. Workspace layout (bytes):
//   [0,384)      p_arr   int[96]
//   [512,896)    w_arr   float[96]
//   [1024,1152)  g_arr   float[32]
//   [4096, +3MB) A       float[96*128*64]
// ---------------------------------------------------------------------------
extern "C" void kernel_launch(void* const* d_in, const int* in_sizes, int n_in,
                              void* d_out, int out_size, void* d_ws, size_t ws_size,
                              hipStream_t stream) {
  const float* x     = (const float*)d_in[0];
  const float* Wdw   = (const float*)d_in[1];
  const float* Wpw   = (const float*)d_in[2];
  const float* gnw   = (const float*)d_in[3];
  const float* gnb   = (const float*)d_in[4];
  const float* Wg    = (const float*)d_in[5];
  const float* gamma = (const float*)d_in[6];
  float* out = (float*)d_out;

  char* ws = (char*)d_ws;
  int*   p_arr = (int*)(ws + 0);
  float* w_arr = (float*)(ws + 512);
  float* g_arr = (float*)(ws + 1024);
  float* A_arr = (float*)(ws + 4096);

  const size_t smem1 = (size_t)(T_ * C_ + 64) * sizeof(float);                 // 262400
  const size_t smem2 = (size_t)(2 * 128 * 65 + 128 + 128 + 64) * sizeof(float)
                     + (size_t)(128 * 66 + 128 * 130) * sizeof(__bf16);        // ~118KB
  const size_t smem3 = (size_t)(3 * 128 * 65 + 2 * 8 * 128 + 128) * sizeof(float);

  k_autocorr<<<B_, 256, smem1, stream>>>(x, p_arr, w_arr, g_arr);
  k_cycconv <<<B_ * K_, 1024, smem2, stream>>>(x, Wdw, Wpw, gnw, gnb, Wg, gamma,
                                               p_arr, A_arr);
  k_combine <<<B_, 1024, smem3, stream>>>(x, A_arr, p_arr, w_arr, g_arr, out);
}
